// AutoregressiveInferenceNet_3444563771467
// MI455X (gfx1250) — matrix operand order repaired and checked by reading, not compiled, not run on hardware
//
#include <hip/hip_runtime.h>

// ---------------- problem dims ----------------
#define B_     64
#define T_     128
#define TS     127        // T-1
#define NNODE  2000
#define E_     300
#define ENC_   256
#define HID_   256
#define HEADS_ 5
#define D_     16
#define IN_    856        // ENC + 2E
#define INP_   864        // padded to 27*32 for K-tiling
#define G3     768        // 3*HID
#define ATT_   1280       // HID*HEADS
#define KT_IH  27         // 864/32
#define KT_HH  8          // 256/32

typedef __attribute__((ext_vector_type(16))) __bf16 v16bf;
typedef __attribute__((ext_vector_type(8)))  float  v8f;

static __device__ __forceinline__ unsigned short f2bf(float f) {
  union { float f; unsigned u; } a; a.f = f;
  unsigned u = a.u;
  u += 0x7fffu + ((u >> 16) & 1u);          // round to nearest even
  return (unsigned short)(u >> 16);
}
static __device__ __forceinline__ float bf2f(unsigned short s) {
  union { unsigned u; float f; } a; a.u = ((unsigned)s) << 16;
  return a.f;
}
static __device__ __forceinline__ v16bf frag_from(uint4 lo, uint4 hi) {
  union { uint4 q[2]; v16bf v; } u; u.q[0] = lo; u.q[1] = hi; return u.v;
}
// A fragment (16x32 bf16, M=row per lane&15): two 16B chunks, K split per ISA 7.12.2
static __device__ __forceinline__ v16bf load_a(const unsigned short* rowPtr, int kt, int lane) {
  int k0 = kt * 32 + ((lane & 16) ? 8 : 0);
  uint4 lo = *(const uint4*)(rowPtr + k0);
  uint4 hi = *(const uint4*)(rowPtr + k0 + 16);
  return frag_from(lo, hi);
}
// B fragment (32x16 bf16): lane n holds two 8-K chunks of column n, K split same as A
static __device__ __forceinline__ v16bf load_b(const unsigned short* rowPtr, int kt, int lane) {
  int k0 = kt * 32 + ((lane & 16) ? 8 : 0);
  uint4 lo = *(const uint4*)(rowPtr + k0);
  uint4 hi = *(const uint4*)(rowPtr + k0 + 16);
  return frag_from(lo, hi);
}
static __device__ __forceinline__ v8f wmma_bf16(v16bf a, v16bf b, v8f c) {
  return __builtin_amdgcn_wmma_f32_16x16x32_bf16(false, a, false, b, (short)0, c, false, false);
}
static __device__ __forceinline__ float fsigmoid(float x) {
  return 1.f / (1.f + __expf(-x));
}
static __device__ __forceinline__ float ftanh(float x) {
  float e2 = __expf(-2.f * x);
  return (1.f - e2) / (1.f + e2);
}

// ---------------- prep kernels ----------------
__global__ void fill_zero_kernel(float* p, long long n) {
  long long i = (long long)blockIdx.x * blockDim.x + threadIdx.x;
  long long stride = (long long)gridDim.x * blockDim.x;
  for (; i < n; i += stride) p[i] = 0.f;
}
__global__ void cvt_wih_kernel(const float* w, unsigned short* o) {
  int i = blockIdx.x * blockDim.x + threadIdx.x;
  int stride = gridDim.x * blockDim.x;
  for (; i < G3 * INP_; i += stride) {
    int r = i / INP_, c = i - r * INP_;
    o[i] = (c < IN_) ? f2bf(w[r * IN_ + c]) : (unsigned short)0;
  }
}
__global__ void cvt_plain_kernel(const float* w, unsigned short* o, int n) {
  int i = blockIdx.x * blockDim.x + threadIdx.x;
  int stride = gridDim.x * blockDim.x;
  for (; i < n; i += stride) o[i] = f2bf(w[i]);
}

// ---------------- persistent scan kernel (one WGP, 32 wave32 waves) ----------------
__global__ __launch_bounds__(1024, 1)
void ar_scan_kernel(const float* __restrict__ program_emb,
                    const float* __restrict__ h0,
                    const float* __restrict__ node_emb,
                    const float* __restrict__ label_emb,
                    const float* __restrict__ b_ih,
                    const float* __restrict__ b_hh,
                    const float* __restrict__ ws2,
                    const float* __restrict__ pred_w,
                    const float* __restrict__ pred_b,
                    const int*   __restrict__ trace,
                    const int*   __restrict__ assign,
                    const unsigned short* __restrict__ w_ih_bf,
                    const unsigned short* __restrict__ w_hh_bf,
                    const unsigned short* __restrict__ ws1_bf,
                    float* __restrict__ storeG,     // [B][TS][HID]
                    float* __restrict__ scacheG,    // [TS][B*HEADS] cached raw scores
                    float* __restrict__ out)        // outs then alphas
{
  // ---- 300.5 KB LDS carve-up (CDNA5: 320KB/WGP) ----
  __shared__ __align__(16) unsigned char smem[307712];
  unsigned short* x_bf   = (unsigned short*)(smem);            // [64][864] bf16 (phase 0-1)
  unsigned short* gi_bf  = (unsigned short*)(smem + 110592);   // [64][768] bf16 (phase 1-2)
  unsigned short* u_bf   = (unsigned short*)(smem);            // [64][256] bf16 (phase 3, reuses x)
  unsigned short* ctx_bf = (unsigned short*)(smem + 32768);    // [64][1280] bf16 (phase 5-6)
  unsigned short* h_bf   = (unsigned short*)(smem + 208896);   // [64][256] bf16
  float*          h_f32  = (float*)         (smem + 241664);   // [64][256] f32
  int*            nI     = (int*)           (smem + 307200);   // [64]
  int*            nIp1   = (int*)           (smem + 307456);   // [64]

  const int tid  = threadIdx.x;
  const int lane = tid & 31;
  const int wave = tid >> 5;
  const int col  = lane & 15;                 // N index inside a 16x16 tile
  const int mrow = (lane & 16) ? 8 : 0;       // M base for the 8 C/D vgprs

  float* alphas = out + (long long)TS * B_ * D_;

  // init hidden state h0
  for (int idx = tid; idx < B_ * HID_; idx += 1024) {
    float v = h0[idx];
    h_f32[idx] = v;
    h_bf[idx]  = f2bf(v);
  }

  for (int i = 0; i < TS; ++i) {
    // ---------- phase 0: gather node ids, build x = [assign_emb|prog|node_emb] in bf16 ----------
    if (tid < B_) {
      nI[tid]   = trace[tid * T_ + i];
      nIp1[tid] = trace[tid * T_ + i + 1];
    }
    __syncthreads();
    for (int idx = tid; idx < B_ * INP_; idx += 1024) {
      int b = idx / INP_, c = idx - b * INP_;
      int node = nI[b];
      float v;
      if (c < E_) {
        int a = (node < NNODE) ? assign[b * NNODE + node] : 0;   // padded-column semantics
        v = label_emb[((long long)node * D_ + a) * E_ + c];
      } else if (c < E_ + ENC_) {
        v = program_emb[b * ENC_ + (c - E_)];
      } else if (c < IN_) {
        v = node_emb[(long long)node * E_ + (c - E_ - ENC_)];
      } else {
        v = 0.f;                                                 // K padding
      }
      x_bf[idx] = f2bf(v);
    }
    __syncthreads();

    // ---------- phase 1: GI = x @ W_ih^T + b_ih  (WMMA bf16) ----------
    // Each wave owns row-tile mt = wave>>3 and 6 column tiles; one LDS A-load
    // feeds 3 back-to-back WMMAs on independent accumulators (XDL pipelining).
    {
      const int mt = wave >> 3;
      const int ntBase = (wave & 7) * 6;
      const unsigned short* arow = x_bf + (mt * 16 + col) * INP_;
      for (int g = 0; g < 2; ++g) {
        const int nt0 = ntBase + g * 3;
        const unsigned short* b0 = w_ih_bf + (long long)((nt0 + 0) * 16 + col) * INP_;
        const unsigned short* b1 = w_ih_bf + (long long)((nt0 + 1) * 16 + col) * INP_;
        const unsigned short* b2 = w_ih_bf + (long long)((nt0 + 2) * 16 + col) * INP_;
        v8f acc0 = {0.f,0.f,0.f,0.f,0.f,0.f,0.f,0.f};
        v8f acc1 = acc0, acc2 = acc0;
        for (int kt = 0; kt < KT_IH; ++kt) {
          v16bf a = load_a(arow, kt, lane);
          acc0 = wmma_bf16(a, load_b(b0, kt, lane), acc0);
          acc1 = wmma_bf16(a, load_b(b1, kt, lane), acc1);
          acc2 = wmma_bf16(a, load_b(b2, kt, lane), acc2);
        }
        #pragma unroll
        for (int q = 0; q < 3; ++q) {
          v8f acc = (q == 0) ? acc0 : (q == 1) ? acc1 : acc2;
          float bias = b_ih[(nt0 + q) * 16 + col];
          #pragma unroll
          for (int j = 0; j < 8; ++j)
            gi_bf[(mt * 16 + mrow + j) * G3 + (nt0 + q) * 16 + col] = f2bf(acc[j] + bias);
        }
      }
    }
    __syncthreads();

    // ---------- phase 2: GH (WMMA) + fused GRU gates; column-triples (r,z,n) per unit ----------
    float hnew[2][8];
    for (int rep = 0; rep < 2; ++rep) {
      int uIdx = wave + rep * 32;               // 0..63 : (mt, ct)
      int mt = uIdx >> 4, ct = uIdx & 15;
      const unsigned short* arow = h_bf + (mt * 16 + col) * HID_;
      const unsigned short* brR  = w_hh_bf + (long long)(      ct * 16 + col) * HID_;
      const unsigned short* brZ  = w_hh_bf + (long long)(256 + ct * 16 + col) * HID_;
      const unsigned short* brN  = w_hh_bf + (long long)(512 + ct * 16 + col) * HID_;
      v8f aR = {0.f,0.f,0.f,0.f,0.f,0.f,0.f,0.f};
      v8f aZ = aR, aN = aR;
      for (int kt = 0; kt < KT_HH; ++kt) {
        v16bf a = load_a(arow, kt, lane);
        aR = wmma_bf16(a, load_b(brR, kt, lane), aR);
        aZ = wmma_bf16(a, load_b(brZ, kt, lane), aZ);
        aN = wmma_bf16(a, load_b(brN, kt, lane), aN);
      }
      float bR = b_hh[      ct * 16 + col];
      float bZ = b_hh[256 + ct * 16 + col];
      float bN = b_hh[512 + ct * 16 + col];
      #pragma unroll
      for (int j = 0; j < 8; ++j) {
        int b = mt * 16 + mrow + j;
        int o = ct * 16 + col;
        float giR = bf2f(gi_bf[b * G3 + o]);
        float giZ = bf2f(gi_bf[b * G3 + 256 + o]);
        float giN = bf2f(gi_bf[b * G3 + 512 + o]);
        float r = fsigmoid(giR + aR[j] + bR);
        float z = fsigmoid(giZ + aZ[j] + bZ);
        float n = ftanh(giN + r * (aN[j] + bN));
        float hp = h_f32[b * HID_ + o];
        hnew[rep][j] = (1.f - z) * n + z * hp;
      }
    }
    __syncthreads();   // all reads of h done before any write
    for (int rep = 0; rep < 2; ++rep) {
      int uIdx = wave + rep * 32;
      int mt = uIdx >> 4, ct = uIdx & 15;
      #pragma unroll
      for (int j = 0; j < 8; ++j) {
        int b = mt * 16 + mrow + j;
        int o = ct * 16 + col;
        float v = hnew[rep][j];
        h_f32[b * HID_ + o] = v;
        h_bf[b * HID_ + o]  = f2bf(v);
        storeG[((long long)b * TS + i) * HID_ + o] = v;
      }
    }
    __syncthreads();

    // ---------- phase 3: incremental attention score for new row only ----------
    // u = tanh(h_new @ ws1^T): wave owns mt = wave>>3, 2 nt tiles sharing one A fragment
    {
      const int mt = wave >> 3;
      const int nt0 = (wave & 7) * 2;
      const unsigned short* arow = h_bf + (mt * 16 + col) * HID_;
      const unsigned short* b0 = ws1_bf + (long long)((nt0 + 0) * 16 + col) * HID_;
      const unsigned short* b1 = ws1_bf + (long long)((nt0 + 1) * 16 + col) * HID_;
      v8f acc0 = {0.f,0.f,0.f,0.f,0.f,0.f,0.f,0.f};
      v8f acc1 = acc0;
      for (int kt = 0; kt < KT_HH; ++kt) {
        v16bf a = load_a(arow, kt, lane);
        acc0 = wmma_bf16(a, load_b(b0, kt, lane), acc0);
        acc1 = wmma_bf16(a, load_b(b1, kt, lane), acc1);
      }
      #pragma unroll
      for (int j = 0; j < 8; ++j) {
        u_bf[(mt * 16 + mrow + j) * HID_ + (nt0 + 0) * 16 + col] = f2bf(ftanh(acc0[j]));
        u_bf[(mt * 16 + mrow + j) * HID_ + (nt0 + 1) * 16 + col] = f2bf(ftanh(acc1[j]));
      }
    }
    __syncthreads();
    // scores_new[b][head] = u[b,:] . ws2[head,:]  -> global score cache row i
    if (tid < B_ * HEADS_) {
      int b = tid / HEADS_, hd = tid - (tid / HEADS_) * HEADS_;
      float s = 0.f;
      for (int k = 0; k < HID_; ++k)
        s += bf2f(u_bf[b * HID_ + k]) * ws2[hd * HID_ + k];
      scacheG[(long long)i * (B_ * HEADS_) + tid] = s;
    }
    __threadfence();
    __syncthreads();

    // ---------- phase 4: softmax over cached scores t<=i -> alphas output ----------
    if (tid < B_ * HEADS_) {
      int b = tid / HEADS_, hd = tid - (tid / HEADS_) * HEADS_;
      float m = -3.0e38f;
      for (int t = 0; t <= i; ++t)
        m = fmaxf(m, scacheG[(long long)t * (B_ * HEADS_) + tid]);
      float sum = 0.f;
      for (int t = 0; t <= i; ++t)
        sum += __expf(scacheG[(long long)t * (B_ * HEADS_) + tid] - m);
      float inv = 1.f / sum;
      long long base = (long long)i * (B_ * TS * HEADS_) + (long long)b * (TS * HEADS_) + hd;
      for (int t = 0; t <= i; ++t)
        alphas[base + (long long)t * HEADS_] =
            __expf(scacheG[(long long)t * (B_ * HEADS_) + tid] - m) * inv;
      // t > i stays exactly 0 from the fill kernel (matches -1e9 mask underflow)
    }
    __threadfence();
    __syncthreads();

    // ---------- phase 5: ctx[b, h*256+d] = sum_t alpha[b,t,h] * store[b,t,d] ----------
    for (int idx = tid; idx < B_ * ATT_; idx += 1024) {
      int b = idx / ATT_;
      int r = idx - b * ATT_;
      int hd = r >> 8;
      int d  = r & 255;
      const float* ap = alphas + (long long)i * (B_ * TS * HEADS_)
                               + (long long)b * (TS * HEADS_) + hd;
      const float* sp = storeG + (long long)b * TS * HID_ + d;
      float acc = 0.f;
      for (int t = 0; t <= i; ++t)
        acc += ap[(long long)t * HEADS_] * sp[(long long)t * HID_];
      ctx_bf[idx] = f2bf(acc);
    }
    __syncthreads();

    // ---------- phase 6: routed expert head (per-row expert, L2-resident table) ----------
    {
      int b = tid >> 4, k = tid & 15;
      int node = nIp1[b];
      const float* pw = pred_w + (long long)node * (ATT_ * D_) + k;
      float acc = pred_b[node * D_ + k];
      const unsigned short* cb = ctx_bf + b * ATT_;
      for (int d = 0; d < ATT_; ++d)
        acc += bf2f(cb[d]) * pw[(long long)d * D_];
      out[(long long)i * (B_ * D_) + tid] = acc;   // tid == b*16+k
    }
    __syncthreads();
  }
}

// ---------------- host-side launch ----------------
extern "C" void kernel_launch(void* const* d_in, const int* in_sizes, int n_in,
                              void* d_out, int out_size, void* d_ws, size_t ws_size,
                              hipStream_t stream) {
  (void)in_sizes; (void)n_in; (void)ws_size;
  const float* program_emb = (const float*)d_in[0];
  const float* h0          = (const float*)d_in[1];
  const float* node_emb    = (const float*)d_in[2];
  const float* label_emb   = (const float*)d_in[3];
  const float* gru_w_ih    = (const float*)d_in[4];
  const float* gru_w_hh    = (const float*)d_in[5];
  const float* gru_b_ih    = (const float*)d_in[6];
  const float* gru_b_hh    = (const float*)d_in[7];
  const float* attn_ws1    = (const float*)d_in[8];
  const float* attn_ws2    = (const float*)d_in[9];
  const float* pred_w      = (const float*)d_in[10];
  const float* pred_b      = (const float*)d_in[11];
  const int*   trace       = (const int*)d_in[12];
  const int*   assign      = (const int*)d_in[13];
  // d_in[14] (execution_lengths) is a no-op in the reference

  float* out = (float*)d_out;

  // workspace carve-up (~10.3 MB)
  unsigned char* ws = (unsigned char*)d_ws;
  size_t off = 0;
  auto carve = [&](size_t bytes) {
    size_t p = off;
    off = (off + bytes + 255) & ~(size_t)255;
    return p;
  };
  unsigned short* w_ih_bf = (unsigned short*)(ws + carve((size_t)G3 * INP_ * 2));
  unsigned short* w_hh_bf = (unsigned short*)(ws + carve((size_t)G3 * HID_ * 2));
  unsigned short* ws1_bf  = (unsigned short*)(ws + carve((size_t)HID_ * HID_ * 2));
  float*          storeG  = (float*)(ws + carve((size_t)B_ * TS * HID_ * 4));
  float*          scacheG = (float*)(ws + carve((size_t)TS * B_ * HEADS_ * 4));

  // 1) zero outputs (alphas t>i must be exact zeros)
  fill_zero_kernel<<<2048, 256, 0, stream>>>(out, (long long)out_size);
  // 2) one-time (per call) bf16 weight conversion for the WMMA GEMMs
  cvt_wih_kernel<<<1024, 256, 0, stream>>>(gru_w_ih, w_ih_bf);
  cvt_plain_kernel<<<512, 256, 0, stream>>>(gru_w_hh, w_hh_bf, G3 * HID_);
  cvt_plain_kernel<<<256, 256, 0, stream>>>(attn_ws1, ws1_bf, HID_ * HID_);
  // 3) persistent single-WGP sequential scan
  ar_scan_kernel<<<1, 1024, 0, stream>>>(program_emb, h0, node_emb, label_emb,
                                         gru_b_ih, gru_b_hh, attn_ws2,
                                         pred_w, pred_b, trace, assign,
                                         w_ih_bf, w_hh_bf, ws1_bf,
                                         storeG, scacheG, out);
}
